// MaskGenWindow_57423712748165
// MI455X (gfx1250) — matrix operand order repaired and checked by reading, not compiled, not run on hardware
//
#include <hip/hip_runtime.h>
#include <hip/hip_bf16.h>

// MI455X / gfx1250 implementation. All matrix math on v_wmma_f32_16x16x32_f16.
typedef __attribute__((ext_vector_type(16))) _Float16 v16h;
typedef __attribute__((ext_vector_type(8)))  float    v8f;

#define T_LEN 1024
#define B_SZ  32
#define D_ZD  128
#define D_SRCD 112
#define NHEADS 4
#define DHEAD 32
#define DFF_D 512
#define TBTOK (T_LEN * B_SZ)

// ---- WMMA fragment index helpers (ISA 7.12.2 layouts, wave32) ----
// A (16x32 f16): lane l holds row M = l&15; element e -> K = (l&16?8:0) + e + (e>=8?8:0)
// B (32x16 f16): lane l holds col N = l&15; element e -> K = (l&16?8:0) + e + (e>=8?8:0)
// C (16x16 f32): VGPR r, lanes 0-15 -> M=r, lanes 16-31 -> M=r+8; N = l&15
static __device__ __forceinline__ int a_kidx(int lane, int e) {
  return ((lane & 16) ? 8 : 0) + e + ((e >= 8) ? 8 : 0);
}
static __device__ __forceinline__ int b_kidx(int lane, int e) {
  return ((lane & 16) ? 8 : 0) + e + ((e >= 8) ? 8 : 0);
}

// ============================ build x = concat(src, pos_enc) ============================
__global__ void k_build_x(const float* __restrict__ src, const float* __restrict__ times,
                          float* __restrict__ x) {
  int i = blockIdx.x * blockDim.x + threadIdx.x;
  if (i >= TBTOK * D_ZD) return;
  int d  = i & (D_ZD - 1);
  int tb = i >> 7;
  float v;
  if (d < D_SRCD) {
    v = src[(size_t)tb * D_SRCD + d];
  } else {
    int j = d - D_SRCD;          // 0..15 ; first 8 sin, next 8 cos
    int jj = j & 7;
    float ts = exp2f(10.0f * (float)jj / 7.0f);   // 1024 ** (jj/7)
    float sv = times[tb] / ts;
    v = (j < 8) ? __sinf(sv) : __cosf(sv);
  }
  x[i] = v;
}

// ============================ generic WMMA GEMM (LDS-staged tiles) ============================
// C[M,N] = act(A[M,K] @ W[K,N] + bias),  act: 0=none 1=relu 2=prelu(alpha)
// Block: 256 threads = 8 waves; tile M=128 (16 rows/wave), N=16, K-step 32.
// A tile staged as f16 in LDS via coalesced b128 loads; W tile staged once per block.
#define A_PAD 34   // row stride in halves; 17-bank step, coprime with 64 -> conflict-free
__global__ void k_gemm(const float* __restrict__ A, const float* __restrict__ W,
                       const float* __restrict__ bias, float* __restrict__ C,
                       int M, int N, int K, int act, const float* __restrict__ alphaPtr) {
  __shared__ _Float16 aT[128][A_PAD];
  __shared__ _Float16 wT[32][16];
  int tid = threadIdx.x;
  int lane = tid & 31;
  int wave = tid >> 5;
  int m0 = blockIdx.x * 128;
  int n0 = blockIdx.y * 16;
  int mw = m0 + wave * 16;
  int ncol = n0 + (lane & 15);

  v8f acc = {};
  for (int k0 = 0; k0 < K; k0 += 32) {
    // ---- stage A tile 128x32 f32 -> f16 (coalesced float4 loads, 4 per thread) ----
#pragma unroll
    for (int i = 0; i < 4; ++i) {
      int idx = tid + 256 * i;               // 0..1023
      int row = idx >> 3;                    // 0..127
      int c4  = (idx & 7) << 2;              // 0,4,...,28
      const float4 f = *(const float4*)&A[(size_t)(m0 + row) * K + k0 + c4];
      aT[row][c4 + 0] = (_Float16)f.x;
      aT[row][c4 + 1] = (_Float16)f.y;
      aT[row][c4 + 2] = (_Float16)f.z;
      aT[row][c4 + 3] = (_Float16)f.w;
      if (k0 + 32 < K)
        __builtin_prefetch(&A[(size_t)(m0 + row) * K + k0 + 32 + c4], 0, 0);
    }
    // ---- stage W tile 32x16 (shared by all 8 waves) ----
#pragma unroll
    for (int i = 0; i < 2; ++i) {
      int idx = tid + 256 * i;               // 0..511
      int kr = idx >> 4;                     // 0..31
      int j  = idx & 15;
      wT[kr][j] = (_Float16)W[(size_t)(k0 + kr) * N + n0 + j];
    }
    __syncthreads();

    v16h afr, bfr;
#pragma unroll
    for (int e = 0; e < 16; ++e) {
      afr[e] = aT[wave * 16 + (lane & 15)][a_kidx(lane, e)];
      bfr[e] = wT[b_kidx(lane, e)][lane & 15];
    }
    acc = __builtin_amdgcn_wmma_f32_16x16x32_f16(false, afr, false, bfr,
                                                 (short)0, acc, false, false);
    __syncthreads();
  }

  float alpha = alphaPtr ? alphaPtr[0] : 0.0f;
  float bv = bias ? bias[ncol] : 0.0f;
  int rbase = mw + ((lane & 16) ? 8 : 0);
#pragma unroll
  for (int r = 0; r < 8; ++r) {
    float v = acc[r] + bv;
    if (act == 1) v = v > 0.0f ? v : 0.0f;
    else if (act == 2) v = v >= 0.0f ? v : alpha * v;
    C[(size_t)(rbase + r) * N + ncol] = v;
  }
}

// ============================ 16-row block softmax over LDS ============================
// sc: [16][Tk] f32 logits -> exp in place; pb: [16][Tk] f16 normalized probs
// red: >= 288 floats. blockDim.x == 256.
static __device__ __forceinline__ void softmax_rows16(float* sc, _Float16* pb,
                                                      float* red, int Tk) {
  int srow = threadIdx.x >> 4;   // 0..15
  int seg  = threadIdx.x & 15;   // 0..15
  int cw = Tk >> 4;
  float m = -1e30f;
  for (int i = seg * cw; i < seg * cw + cw; ++i) m = fmaxf(m, sc[srow * Tk + i]);
  red[srow * 16 + seg] = m;
  __syncthreads();
  if (seg == 0) {
    float mm = -1e30f;
    for (int j = 0; j < 16; ++j) mm = fmaxf(mm, red[srow * 16 + j]);
    red[256 + srow] = mm;
  }
  __syncthreads();
  float rm = red[256 + srow];
  float s = 0.0f;
  for (int i = seg * cw; i < seg * cw + cw; ++i) {
    float e = __expf(sc[srow * Tk + i] - rm);
    sc[srow * Tk + i] = e;
    s += e;
  }
  red[srow * 16 + seg] = s;
  __syncthreads();
  if (seg == 0) {
    float ss = 0.0f;
    for (int j = 0; j < 16; ++j) ss += red[srow * 16 + j];
    red[272 + srow] = ss;
  }
  __syncthreads();
  float rinv = 1.0f / red[272 + srow];
  for (int i = seg * cw; i < seg * cw + cw; ++i)
    pb[srow * Tk + i] = (_Float16)(sc[srow * Tk + i] * rinv);
  __syncthreads();
}

// ============================ fused attention (per b,h, 16-row q tile) ============================
// q,k,v: (T, B, 128) f32 with head h in cols [h*32, h*32+32). out: same layout.
__global__ void k_attention(const float* __restrict__ q, const float* __restrict__ kk,
                            const float* __restrict__ vv, float* __restrict__ out,
                            int Tk, float scale) {
  extern __shared__ char smem[];
  float*    sc  = (float*)smem;                                 // 16*Tk f32
  _Float16* pb  = (_Float16*)(smem + (size_t)16 * Tk * 4);      // 16*Tk f16
  float*    red = (float*)(smem + (size_t)16 * Tk * 4 + (size_t)16 * Tk * 2); // 288 f32

  int t0 = blockIdx.x * 16;
  int bh = blockIdx.y;
  int b = bh >> 2;
  int h = bh & 3;
  int lane = threadIdx.x & 31;
  int wave = threadIdx.x >> 5;
  size_t hoff = (size_t)b * D_ZD + (size_t)h * DHEAD;

  // A-frag: 16 q rows x 32 head dims (loaded once)
  v16h afr;
#pragma unroll
  for (int e = 0; e < 16; ++e) {
    int ka = a_kidx(lane, e);
    afr[e] = (_Float16)q[(size_t)(t0 + (lane & 15)) * (B_SZ * D_ZD) + hoff + ka];
  }

  // phase 1: scores = q @ k^T (each wave strides over kv tiles)
  for (int kt = wave; kt < (Tk >> 4); kt += 8) {
    v16h bfr;
#pragma unroll
    for (int e = 0; e < 16; ++e) {
      int kd = b_kidx(lane, e);
      int tk = kt * 16 + (lane & 15);
      bfr[e] = (_Float16)kk[(size_t)tk * (B_SZ * D_ZD) + hoff + kd];
    }
    v8f c = {};
    c = __builtin_amdgcn_wmma_f32_16x16x32_f16(false, afr, false, bfr,
                                               (short)0, c, false, false);
#pragma unroll
    for (int r = 0; r < 8; ++r) {
      int rr = r + ((lane & 16) ? 8 : 0);
      sc[rr * Tk + kt * 16 + (lane & 15)] = c[r] * scale;
    }
  }
  __syncthreads();

  // phase 2: softmax -> f16 probs
  softmax_rows16(sc, pb, red, Tk);

  // phase 3: out = P @ V; each wave owns a K-range, partials reduced in LDS
  v8f acc0 = {}, acc1 = {};
  int per = (Tk >> 5) >> 3;   // kv chunks of 32 per wave
  for (int cc = wave * per; cc < wave * per + per; ++cc) {
    v16h pa, b0, b1;
#pragma unroll
    for (int e = 0; e < 16; ++e) {
      int kva = cc * 32 + a_kidx(lane, e);
      pa[e] = pb[(lane & 15) * Tk + kva];
      int kvb = cc * 32 + b_kidx(lane, e);
      size_t vo = (size_t)kvb * (B_SZ * D_ZD) + hoff;
      b0[e] = (_Float16)vv[vo + (lane & 15)];
      b1[e] = (_Float16)vv[vo + 16 + (lane & 15)];
    }
    acc0 = __builtin_amdgcn_wmma_f32_16x16x32_f16(false, pa, false, b0,
                                                  (short)0, acc0, false, false);
    acc1 = __builtin_amdgcn_wmma_f32_16x16x32_f16(false, pa, false, b1,
                                                  (short)0, acc1, false, false);
  }
  float* part = sc;  // reuse score region: [wave][16][32]
#pragma unroll
  for (int r = 0; r < 8; ++r) {
    int rr = r + ((lane & 16) ? 8 : 0);
    part[(wave * 16 + rr) * 32 + (lane & 15)]      = acc0[r];
    part[(wave * 16 + rr) * 32 + 16 + (lane & 15)] = acc1[r];
  }
  __syncthreads();
  for (int i = threadIdx.x; i < 16 * 32; i += 256) {
    int rr = i >> 5, col = i & 31;
    float s = 0.0f;
    for (int w = 0; w < 8; ++w) s += part[(w * 16 + rr) * 32 + col];
    out[(size_t)(t0 + rr) * (B_SZ * D_ZD) + hoff + col] = s;
  }
}

// ============================ residual + layernorm (per token) ============================
__global__ void k_addln(const float* __restrict__ x, const float* __restrict__ h,
                        const float* __restrict__ g, const float* __restrict__ bb,
                        float* __restrict__ y) {
  __shared__ float red[128];
  __shared__ float mv, varv;
  int tok = blockIdx.x;
  int d = threadIdx.x;
  size_t base = (size_t)tok * D_ZD;
  float v = x[base + d] + h[base + d];
  red[d] = v;
  __syncthreads();
  for (int s = 64; s > 0; s >>= 1) { if (d < s) red[d] += red[d + s]; __syncthreads(); }
  if (d == 0) mv = red[0] * (1.0f / 128.0f);
  __syncthreads();
  float c = v - mv;
  red[d] = c * c;
  __syncthreads();
  for (int s = 64; s > 0; s >>= 1) { if (d < s) red[d] += red[d + s]; __syncthreads(); }
  if (d == 0) varv = red[0] * (1.0f / 128.0f);
  __syncthreads();
  y[base + d] = c * rsqrtf(varv + 1e-5f) * g[d] + bb[d];
}

// ============================ N==1 projection: out[b,t] = x[t,b,:].W + bias ============================
__global__ void k_dotcol(const float* __restrict__ x, const float* __restrict__ w,
                         const float* __restrict__ bias, float* __restrict__ out) {
  int i = blockIdx.x * blockDim.x + threadIdx.x;   // token = t*B + b
  if (i >= TBTOK) return;
  int t = i / B_SZ, b = i % B_SZ;
  float s = bias[0];
  for (int d = 0; d < D_ZD; ++d) s += x[(size_t)i * D_ZD + d] * w[d];
  out[(size_t)b * T_LEN + t] = s;
}

// ============================ row softmax over T (grid = B) ============================
__global__ void k_softmax_rows(const float* __restrict__ in, float* __restrict__ out) {
  __shared__ float red[256];
  __shared__ float stat;
  int b = blockIdx.x, tid = threadIdx.x;
  float m = -1e30f;
  for (int t = tid; t < T_LEN; t += 256) m = fmaxf(m, in[(size_t)b * T_LEN + t]);
  red[tid] = m; __syncthreads();
  for (int s = 128; s > 0; s >>= 1) { if (tid < s) red[tid] = fmaxf(red[tid], red[tid + s]); __syncthreads(); }
  if (tid == 0) stat = red[0];
  __syncthreads();
  float mm = stat;
  float sum = 0.0f;
  for (int t = tid; t < T_LEN; t += 256) sum += __expf(in[(size_t)b * T_LEN + t] - mm);
  red[tid] = sum; __syncthreads();
  for (int s = 128; s > 0; s >>= 1) { if (tid < s) red[tid] += red[tid + s]; __syncthreads(); }
  if (tid == 0) stat = red[0];
  __syncthreads();
  float inv = 1.0f / stat;
  for (int t = tid; t < T_LEN; t += 256)
    out[(size_t)b * T_LEN + t] = __expf(in[(size_t)b * T_LEN + t] - mm) * inv;
}

// ============================ gumbel hard one-hot (forward value == one-hot) ============================
static __device__ __forceinline__ unsigned hmix(unsigned x) {
  x ^= x >> 16; x *= 0x7feb352dU; x ^= x >> 15; x *= 0x846ca68bU; x ^= x >> 16; return x;
}
__global__ void k_gumbel_onehot(const float* __restrict__ phi, float* __restrict__ out,
                                unsigned seed) {
  __shared__ float rv[256];
  __shared__ int ri[256];
  int b = blockIdx.x, tid = threadIdx.x;
  float best = -1e30f; int bi = 0;
  for (int t = tid; t < T_LEN; t += 256) {
    unsigned hv = hmix(seed ^ ((unsigned)(b * T_LEN + t) * 2654435761U));
    float u = ((float)hv + 0.5f) * (1.0f / 4294967296.0f);
    u = fminf(fmaxf(u, 1e-10f), 0.99999994f);
    float gmb = -__logf(-__logf(u));
    float val = __logf(phi[(size_t)b * T_LEN + t] + 1e-9f) + gmb;
    if (val > best) { best = val; bi = t; }
  }
  rv[tid] = best; ri[tid] = bi;
  __syncthreads();
  for (int s = 128; s > 0; s >>= 1) {
    if (tid < s && rv[tid + s] > rv[tid]) { rv[tid] = rv[tid + s]; ri[tid] = ri[tid + s]; }
    __syncthreads();
  }
  int arg = ri[0];
  for (int t = tid; t < T_LEN; t += 256)
    out[(size_t)b * T_LEN + t] = (t == arg) ? 1.0f : 0.0f;
}

// ============================ full_mask: cL*sR + cR*sL (per batch serial scan) ============================
__global__ void k_fullmask(const float* __restrict__ pL, const float* __restrict__ pR,
                           float* __restrict__ out) {
  int b = threadIdx.x;
  if (b >= B_SZ) return;
  const float* l = pL + (size_t)b * T_LEN;
  const float* r = pR + (size_t)b * T_LEN;
  float totL = 0.0f, totR = 0.0f;
  for (int t = 0; t < T_LEN; ++t) { totL += l[t]; totR += r[t]; }
  float cL = 0.0f, cR = 0.0f;
  for (int t = 0; t < T_LEN; ++t) {
    cL += l[t]; cR += r[t];
    float sL = totL - cL + l[t];
    float sR = totR - cR + r[t];
    out[(size_t)b * T_LEN + t] = cL * sR + cR * sL;
  }
}

// ============================ column max over T: (T*B,32) -> (B,32) ============================
__global__ void k_colmax(const float* __restrict__ z, float* __restrict__ agg) {
  int i = blockIdx.x * blockDim.x + threadIdx.x;
  if (i >= B_SZ * 32) return;
  int b = i >> 5, j = i & 31;
  float m = -1e30f;
  for (int t = 0; t < T_LEN; ++t)
    m = fmaxf(m, z[((size_t)t * B_SZ + b) * 32 + j]);
  agg[i] = m;
}

// ============================ trend head: p = sigmoid(prelu(agg@Wt1+bt1)@Wt2+bt2) ============================
__global__ void k_head(const float* __restrict__ agg, const float* __restrict__ Wt1,
                       const float* __restrict__ bt1, const float* __restrict__ at,
                       const float* __restrict__ Wt2, const float* __restrict__ bt2,
                       float* __restrict__ pbuf, float* __restrict__ pout) {
  __shared__ float sh[B_SZ][32];
  int tid = threadIdx.x;
  if (tid < B_SZ * 32) {
    int b = tid >> 5, n = tid & 31;
    float s = bt1[n];
    for (int k2 = 0; k2 < 32; ++k2) s += agg[b * 32 + k2] * Wt1[k2 * 32 + n];
    float al = at[0];
    sh[b][n] = (s >= 0.0f) ? s : al * s;
  }
  __syncthreads();
  if (tid < B_SZ) {
    float s = bt2[0];
    for (int n = 0; n < 32; ++n) s += sh[tid][n] * Wt2[n];
    float pv = 1.0f / (1.0f + __expf(-s));
    pbuf[tid] = pv;
    pout[tid] = pv;
  }
}

// ============================ exponential smoother: softmax_s(-|dt|/p) @ src ============================
__global__ void k_smoother(const float* __restrict__ times, const float* __restrict__ src,
                           const float* __restrict__ pbuf, float* __restrict__ out) {
  extern __shared__ char smem[];
  const int Tk = T_LEN;
  float*    sc  = (float*)smem;
  _Float16* pb  = (_Float16*)(smem + (size_t)16 * Tk * 4);
  float*    red = (float*)(smem + (size_t)16 * Tk * 4 + (size_t)16 * Tk * 2);
  int t0 = blockIdx.x * 16;
  int b  = blockIdx.y;
  int lane = threadIdx.x & 31;
  int wave = threadIdx.x >> 5;
  float inv_p = 1.0f / fmaxf(pbuf[b], 1e-6f);

  for (int i = threadIdx.x; i < 16 * Tk; i += 256) {
    int rr = i >> 10, s = i & (Tk - 1);
    float dt = fabsf(times[(size_t)(t0 + rr) * B_SZ + b] - times[(size_t)s * B_SZ + b]);
    sc[i] = -dt * inv_p;
  }
  __syncthreads();
  softmax_rows16(sc, pb, red, Tk);

  if (wave < 7) {   // 7 * 16 = 112 output dims
    v8f acc = {};
    for (int cc = 0; cc < (Tk >> 5); ++cc) {
      v16h pa, bfr;
#pragma unroll
      for (int e = 0; e < 16; ++e) {
        int sa = cc * 32 + a_kidx(lane, e);
        pa[e] = pb[(lane & 15) * Tk + sa];
        int sb = cc * 32 + b_kidx(lane, e);
        bfr[e] = (_Float16)src[((size_t)sb * B_SZ + b) * D_SRCD + wave * 16 + (lane & 15)];
      }
      acc = __builtin_amdgcn_wmma_f32_16x16x32_f16(false, pa, false, bfr,
                                                   (short)0, acc, false, false);
    }
#pragma unroll
    for (int r = 0; r < 8; ++r) {
      int rr = r + ((lane & 16) ? 8 : 0);
      out[((size_t)(t0 + rr) * B_SZ + b) * D_SRCD + wave * 16 + (lane & 15)] = acc[r];
    }
  }
}

// ============================================================================
extern "C" void kernel_launch(void* const* d_in, const int* in_sizes, int n_in,
                              void* d_out, int out_size, void* d_ws, size_t ws_size,
                              hipStream_t stream) {
  (void)in_sizes; (void)n_in; (void)out_size; (void)ws_size;
  auto P = [&](int i) -> const float* { return (const float*)d_in[i]; };
  // setup_inputs order: z_seq, src, times, params (jax pytree sorted-key flatten)
  const float* z_seq = P(0);
  const float* src   = P(1);
  const float* times = P(2);
  const float* WL = P(3);  const float* WR = P(4);
  const float* Wa1 = P(5); const float* Wa2 = P(6);
  const float* Wt1 = P(7); const float* Wt2 = P(8);
  const float* a1 = P(9);  const float* a2 = P(10); const float* at = P(11);
  const float* bL = P(12); const float* bR = P(13);
  const float* ba1 = P(14); const float* ba2 = P(15);
  const float* bt1 = P(16); const float* bt2 = P(17);
  const int LBASE = 18, LSTRIDE = 26;
  // per-layer leaves (sorted): W1 W2 b1 b2 | ca{Wk Wo Wq Wv bk bo bq bv} |
  //                            n1b n1g n2b n2g n3b n3g | sa{Wk Wo Wq Wv bk bo bq bv}

  // ---- workspace carve-out (all f32) ----
  char* ws = (char*)d_ws;
  size_t off = 0;
  auto take = [&](size_t bytes) -> float* {
    float* p = (float*)(ws + off);
    off = (off + bytes + 255) & ~(size_t)255;
    return p;
  };
  float* x    = take((size_t)TBTOK * D_ZD * 4);
  float* q    = take((size_t)TBTOK * D_ZD * 4);
  float* k    = take((size_t)TBTOK * D_ZD * 4);
  float* v    = take((size_t)TBTOK * D_ZD * 4);
  float* attn = take((size_t)TBTOK * D_ZD * 4);
  float* o    = take((size_t)TBTOK * D_ZD * 4);
  float* h1   = take((size_t)TBTOK * DFF_D * 4);
  float* zp1  = take((size_t)TBTOK * 32 * 4);
  float* zp2  = take((size_t)TBTOK * 32 * 4);
  float* logL = take((size_t)B_SZ * T_LEN * 4);
  float* logR = take((size_t)B_SZ * T_LEN * 4);
  float* phiL = take((size_t)B_SZ * T_LEN * 4);
  float* phiR = take((size_t)B_SZ * T_LEN * 4);
  float* hotL = take((size_t)B_SZ * T_LEN * 4);
  float* hotR = take((size_t)B_SZ * T_LEN * 4);
  float* agg  = take((size_t)B_SZ * 32 * 4);
  float* pbuf = take((size_t)B_SZ * 4);

  float* out_f = (float*)d_out;
  float* out_smooth_src  = out_f;                                      // (T,B,112)
  float* out_smooth_mask = out_f + (size_t)T_LEN * B_SZ * D_SRCD;      // (B,T)
  float* out_hard_mask   = out_smooth_mask + (size_t)B_SZ * T_LEN;     // (B,T)
  float* out_p           = out_hard_mask + (size_t)B_SZ * T_LEN;       // (B,1)

  auto gemm = [&](const float* A, const float* W, const float* bias, float* C,
                  int M, int N, int K, int act, const float* alpha) {
    dim3 g((unsigned)(M / 128), (unsigned)(N / 16));
    k_gemm<<<g, 256, 0, stream>>>(A, W, bias, C, M, N, K, act, alpha);
  };
  const size_t att_smem = (size_t)16 * 1024 * 4 + (size_t)16 * 1024 * 2 + 288 * 4;
  auto attention = [&](const float* qq, const float* kk2, const float* vv2, float* oo) {
    dim3 g(T_LEN / 16, B_SZ * NHEADS);
    k_attention<<<g, 256, att_smem, stream>>>(qq, kk2, vv2, oo, 1024, 0.17677669529663687f);
  };

  // ---- x = concat(src, pos_enc(times)) ----
  k_build_x<<<(TBTOK * D_ZD + 255) / 256, 256, 0, stream>>>(src, times, x);

  // ---- transformer layers ----
  for (int l = 0; l < 2; ++l) {
    int base = LBASE + LSTRIDE * l;
    const float *W1 = P(base + 0), *W2 = P(base + 1), *b1 = P(base + 2), *b2 = P(base + 3);
    const float *caWk = P(base + 4), *caWo = P(base + 5), *caWq = P(base + 6), *caWv = P(base + 7);
    const float *cabk = P(base + 8), *cabo = P(base + 9), *cabq = P(base + 10), *cabv = P(base + 11);
    const float *n1b = P(base + 12), *n1g = P(base + 13);
    const float *n2b = P(base + 14), *n2g = P(base + 15);
    const float *n3b = P(base + 16), *n3g = P(base + 17);
    const float *saWk = P(base + 18), *saWo = P(base + 19), *saWq = P(base + 20), *saWv = P(base + 21);
    const float *sabk = P(base + 22), *sabo = P(base + 23), *sabq = P(base + 24), *sabv = P(base + 25);

    // self-attention
    gemm(x, saWq, sabq, q, TBTOK, D_ZD, D_ZD, 0, nullptr);
    gemm(x, saWk, sabk, k, TBTOK, D_ZD, D_ZD, 0, nullptr);
    gemm(x, saWv, sabv, v, TBTOK, D_ZD, D_ZD, 0, nullptr);
    attention(q, k, v, attn);
    gemm(attn, saWo, sabo, o, TBTOK, D_ZD, D_ZD, 0, nullptr);
    k_addln<<<TBTOK, 128, 0, stream>>>(x, o, n1g, n1b, x);

    // cross-attention (kv from z_seq)
    gemm(x, caWq, cabq, q, TBTOK, D_ZD, D_ZD, 0, nullptr);
    gemm(z_seq, caWk, cabk, k, TBTOK, D_ZD, D_ZD, 0, nullptr);
    gemm(z_seq, caWv, cabv, v, TBTOK, D_ZD, D_ZD, 0, nullptr);
    attention(q, k, v, attn);
    gemm(attn, caWo, cabo, o, TBTOK, D_ZD, D_ZD, 0, nullptr);
    k_addln<<<TBTOK, 128, 0, stream>>>(x, o, n2g, n2b, x);

    // feed-forward
    gemm(x, W1, b1, h1, TBTOK, DFF_D, D_ZD, 1, nullptr);
    gemm(h1, W2, b2, o, TBTOK, D_ZD, DFF_D, 0, nullptr);
    k_addln<<<TBTOK, 128, 0, stream>>>(x, o, n3g, n3b, x);
  }

  // ---- heads ----
  gemm(x, Wa1, ba1, zp1, TBTOK, 32, D_ZD, 2, a1);
  gemm(zp1, Wa2, ba2, zp2, TBTOK, 32, 32, 2, a2);

  k_dotcol<<<(TBTOK + 255) / 256, 256, 0, stream>>>(x, WL, bL, logL);
  k_dotcol<<<(TBTOK + 255) / 256, 256, 0, stream>>>(x, WR, bR, logR);
  k_softmax_rows<<<B_SZ, 256, 0, stream>>>(logL, phiL);
  k_softmax_rows<<<B_SZ, 256, 0, stream>>>(logR, phiR);
  k_gumbel_onehot<<<B_SZ, 256, 0, stream>>>(phiL, hotL, 0x1234u);
  k_gumbel_onehot<<<B_SZ, 256, 0, stream>>>(phiR, hotR, 0x8765u);
  k_fullmask<<<1, 32, 0, stream>>>(phiL, phiR, out_smooth_mask);
  k_fullmask<<<1, 32, 0, stream>>>(hotL, hotR, out_hard_mask);

  k_colmax<<<(B_SZ * 32 + 255) / 256, 256, 0, stream>>>(zp2, agg);
  k_head<<<1, 1024, 0, stream>>>(agg, Wt1, bt1, at, Wt2, bt2, pbuf, out_p);

  // ---- exponential smoother ----
  {
    dim3 g(T_LEN / 16, B_SZ);
    k_smoother<<<g, 256, att_smem, stream>>>(times, src, pbuf, out_smooth_src);
  }
}
